// Net_6339371729707
// MI455X (gfx1250) — matrix operand order repaired and checked by reading, not compiled, not run on hardware
//
#include <hip/hip_runtime.h>
#include <hip/hip_bf16.h>
#include <math.h>

#define FL 48

typedef __attribute__((ext_vector_type(2)))  float    v2f;
typedef __attribute__((ext_vector_type(8)))  float    v8f;
typedef __attribute__((ext_vector_type(16))) _Float16 v16h;

#if __has_builtin(__builtin_amdgcn_wmma_f32_16x16x4_f32)
#define USE_F32_WMMA 1
#endif

// ---------------------------------------------------------------- utilities

__global__ void fill_zero(float* __restrict__ p, int n) {
    int i = blockIdx.x * blockDim.x + threadIdx.x;
    if (i < n) p[i] = 0.0f;
}

// scatter:  agg[dst[e]*C + c] += x[src[e]*C + c];  cnt[dst[e]] += 1 (c==0)
__global__ void scatter_add(const float* __restrict__ x, const int* __restrict__ eidx,
                            int E, int C, float* __restrict__ agg, float* __restrict__ cnt) {
    const int* src = eidx;
    const int* dst = eidx + E;
    int i = blockIdx.x * blockDim.x + threadIdx.x;
    int total = E * C;
    if (i >= total) return;
    int e = i / C, c = i - e * C;
    int s = src[e], d = dst[e];
    atomicAdd(&agg[d * C + c], x[s * C + c]);
    if (c == 0) atomicAdd(&cnt[d], 1.0f);
}

// level-1 SAGE, C_in = 1: out[n][f] = relu(mean_n*Wl[f] + bl[f] + x_n*Wr[f])
__global__ void sage1(const float* __restrict__ x, const float* __restrict__ agg,
                      const float* __restrict__ cnt, const float* __restrict__ Wl,
                      const float* __restrict__ bl, const float* __restrict__ Wr,
                      float* __restrict__ out) {
    int i = blockIdx.x * blockDim.x + threadIdx.x;
    if (i >= 16384 * FL) return;
    int n = i / FL, f = i - n * FL;
    float mean = agg[n] / fmaxf(cnt[n], 1.0f);
    float v = mean * Wl[f] + bl[f] + x[n] * Wr[f];
    out[i] = fmaxf(v, 0.0f);
}

__global__ void mean_kernel(const float* __restrict__ agg, const float* __restrict__ cnt,
                            float* __restrict__ mean, int N) {
    int i = blockIdx.x * blockDim.x + threadIdx.x;
    if (i >= N * FL) return;
    int n = i / FL;
    mean[i] = agg[i] / fmaxf(cnt[n], 1.0f);
}

// 2x2 max pool on HxH grid of (H*H, FL) node features
__global__ void pool2(const float* __restrict__ in, float* __restrict__ out, int Hout) {
    int i = blockIdx.x * blockDim.x + threadIdx.x;
    int total = Hout * Hout * FL;
    if (i >= total) return;
    int c = i % FL;
    int n = i / FL;
    int r = n / Hout, q = n - r * Hout;
    int Hin = Hout * 2;
    float v = -3.402823e38f;
    #pragma unroll
    for (int di = 0; di < 2; ++di)
        #pragma unroll
        for (int dj = 0; dj < 2; ++dj)
            v = fmaxf(v, in[((2 * r + di) * Hin + (2 * q + dj)) * FL + c]);
    out[i] = v;
}

// ------------------------------------------------ WMMA SAGE projection (levels 2..4)
// out = act([mean | x] (Nx96) @ [Wl ; Wr] (96x48) + bl), one wave = one 16x16 tile.
__global__ __launch_bounds__(32) void sage_gemm_wmma(
    const float* __restrict__ mean, const float* __restrict__ x,
    const float* __restrict__ Wl, const float* __restrict__ Wr,
    const float* __restrict__ bl, float* __restrict__ out, int N, int relu) {
    const int lane = threadIdx.x & 31;
    const int m = lane & 15;     // row (A) / col (B,C) within tile
    const int g = lane >> 4;     // K-group select
    const int row0 = blockIdx.x * 16;
    const int col0 = blockIdx.y * 16;

    v8f acc = {0.f, 0.f, 0.f, 0.f, 0.f, 0.f, 0.f, 0.f};

#if USE_F32_WMMA
    // V_WMMA_F32_16X16X4_F32: A 16x4 f32 (2 VGPR), B 4x16 f32 (2 VGPR), C 16x16 f32
    #pragma unroll
    for (int k = 0; k < 2 * FL; k += 4) {
        const float* Arow;
        const float* B;
        int kk;
        if (k < FL) { Arow = mean + (row0 + m) * FL; B = Wl; kk = k; }
        else        { Arow = x    + (row0 + m) * FL; B = Wr; kk = k - FL; }
        v2f a, b;
        a.x = Arow[kk + 2 * g + 0];                       // lanes 0-15: K=k,k+1 ; 16-31: K=k+2,k+3
        a.y = Arow[kk + 2 * g + 1];
        b.x = B[(kk + 2 * g + 0) * FL + col0 + m];        // B row K, col n
        b.y = B[(kk + 2 * g + 1) * FL + col0 + m];
        acc = __builtin_amdgcn_wmma_f32_16x16x4_f32(false, a, false, b,
                                                    (short)0, acc, false, false);
    }
#else
    // fallback: f32->f16 convert on load, v_wmma_f32_16x16x32_f16 (codegen-confirmed)
    for (int k = 0; k < 2 * FL; k += 32) {
        v16h a, b;
        #pragma unroll
        for (int j = 0; j < 8; ++j) {
            int k1 = k + g * 8 + j;          // halves 0..7
            int k2 = k + 16 + g * 8 + j;     // halves 8..15
            float a1 = (k1 < FL) ? mean[(row0 + m) * FL + k1] : x[(row0 + m) * FL + k1 - FL];
            float a2 = (k2 < FL) ? mean[(row0 + m) * FL + k2] : x[(row0 + m) * FL + k2 - FL];
            a[j]     = (_Float16)a1;
            a[8 + j] = (_Float16)a2;
        }
        #pragma unroll
        for (int j = 0; j < 16; ++j) {
            int kb = k + ((j < 8) ? (g * 8 + j) : (16 + g * 8 + j - 8));
            float bv = (kb < FL) ? Wl[kb * FL + col0 + m] : Wr[(kb - FL) * FL + col0 + m];
            b[j] = (_Float16)bv;
        }
        acc = __builtin_amdgcn_wmma_f32_16x16x32_f16(false, a, false, b,
                                                     (short)0, acc, false, false);
    }
#endif

    #pragma unroll
    for (int r = 0; r < 8; ++r) {            // D: lane n = m, M = r + 8*g
        int M = r + 8 * g;
        float v = acc[r] + bl[col0 + m];
        if (relu) v = fmaxf(v, 0.0f);
        out[(row0 + M) * FL + col0 + m] = v;
    }
}

// ------------------------------------------------ attention block pieces

// per-column mean & max over N rows
__global__ __launch_bounds__(128) void col_reduce(const float* __restrict__ x, int N,
                                                  float* __restrict__ meanv, float* __restrict__ maxv) {
    __shared__ float ssum[128], smax[128];
    int c = blockIdx.x, t = threadIdx.x;
    float s = 0.0f, mx = -3.402823e38f;
    for (int n = t; n < N; n += 128) {
        float v = x[n * FL + c];
        s += v;
        mx = fmaxf(mx, v);
    }
    ssum[t] = s; smax[t] = mx;
    __syncthreads();
    for (int off = 64; off > 0; off >>= 1) {
        if (t < off) { ssum[t] += ssum[t + off]; smax[t] = fmaxf(smax[t], smax[t + off]); }
        __syncthreads();
    }
    if (t == 0) { meanv[c] = ssum[0] / (float)N; maxv[c] = smax[0]; }
}

// att[f] = sigmoid(meanv @ amW + amb + maxv @ axW + axb)
__global__ void att_gate(const float* __restrict__ meanv, const float* __restrict__ maxv,
                         const float* __restrict__ amW, const float* __restrict__ amb,
                         const float* __restrict__ axW, const float* __restrict__ axb,
                         float* __restrict__ att) {
    int f = threadIdx.x;
    if (f >= FL) return;
    float s = amb[f] + axb[f];
    for (int c = 0; c < FL; ++c)
        s += meanv[c] * amW[c * FL + f] + maxv[c] * axW[c * FL + f];
    att[f] = 1.0f / (1.0f + expf(-s));
}

// ch[n] = sigmoid(mean_n @ cWl + cbl + x_n @ cWr)
__global__ void ch_gate(const float* __restrict__ x, const float* __restrict__ agg,
                        const float* __restrict__ cnt, const float* __restrict__ cWl,
                        const float* __restrict__ cbl, const float* __restrict__ cWr,
                        float* __restrict__ ch, int N) {
    int n = blockIdx.x * blockDim.x + threadIdx.x;
    if (n >= N) return;
    float inv = 1.0f / fmaxf(cnt[n], 1.0f);
    float s = cbl[0];
    for (int c = 0; c < FL; ++c)
        s += agg[n * FL + c] * inv * cWl[c] + x[n * FL + c] * cWr[c];
    ch[n] = 1.0f / (1.0f + expf(-s));
}

// x = x * (1 + att[c] + ch[n])   (in place)
__global__ void att_apply(float* __restrict__ x, const float* __restrict__ att,
                          const float* __restrict__ ch, int N) {
    int i = blockIdx.x * blockDim.x + threadIdx.x;
    if (i >= N * FL) return;
    int n = i / FL, c = i - n * FL;
    x[i] = x[i] * (1.0f + att[c] + ch[n]);
}

// ------------------------------------------------ head

// h[o] = relu(sum_k flat[k]*W[k][o] + b[o]); flat[c*256+n] = x4[n][c]
__global__ __launch_bounds__(128) void fc1_kernel(const float* __restrict__ x4,
                                                  const float* __restrict__ W,
                                                  const float* __restrict__ b,
                                                  float* __restrict__ h) {
    __shared__ float red[128];
    int o = blockIdx.x, t = threadIdx.x;
    float s = 0.0f;
    for (int k = t; k < FL * 256; k += 128) {
        int n = k & 255;
        int c = k >> 8;
        s += x4[n * FL + c] * W[k * 256 + o];
    }
    red[t] = s;
    __syncthreads();
    for (int off = 64; off > 0; off >>= 1) {
        if (t < off) red[t] += red[t + off];
        __syncthreads();
    }
    if (t == 0) h[o] = fmaxf(red[0] + b[o], 0.0f);
}

__global__ void fc2_logsoftmax(const float* __restrict__ h, const float* __restrict__ W,
                               const float* __restrict__ b, float* __restrict__ out) {
    __shared__ float logits[10];
    int t = threadIdx.x;
    if (t < 10) {
        float s = b[t];
        for (int k = 0; k < 256; ++k) s += h[k] * W[k * 10 + t];
        logits[t] = s;
    }
    __syncthreads();
    if (t < 10) {
        float mx = logits[0];
        for (int i = 1; i < 10; ++i) mx = fmaxf(mx, logits[i]);
        float se = 0.0f;
        for (int i = 0; i < 10; ++i) se += expf(logits[i] - mx);
        out[t] = logits[t] - mx - logf(se);
    }
}

// ---------------------------------------------------------------- host side

static inline int gb(int n) { return (n + 255) / 256; }

extern "C" void kernel_launch(void* const* d_in, const int* in_sizes, int n_in,
                              void* d_out, int out_size, void* d_ws, size_t ws_size,
                              hipStream_t stream) {
    // ---- inputs (setup_inputs dict order) ----
    const float* x_image = (const float*)d_in[0];
    const int*   e0 = (const int*)d_in[1];   // (2, 65536)
    const int*   e1 = (const int*)d_in[2];   // (2, 16384)
    const int*   e2 = (const int*)d_in[3];   // (2, 4096)
    const int*   e3 = (const int*)d_in[4];   // (2, 1024)
    // d_in[5] = train (unused)
    const float* gWl[4] = {(const float*)d_in[6],  (const float*)d_in[9],
                           (const float*)d_in[12], (const float*)d_in[15]};
    const float* gbl[4] = {(const float*)d_in[7],  (const float*)d_in[10],
                           (const float*)d_in[13], (const float*)d_in[16]};
    const float* gWr[4] = {(const float*)d_in[8],  (const float*)d_in[11],
                           (const float*)d_in[14], (const float*)d_in[17]};
    // attention blocks: 7 tensors each, starting at 18 (block 4 unused by reference)
    const float* amW[3]; const float* amb[3]; const float* axW[3]; const float* axb[3];
    const float* cWl[3]; const float* cbl[3]; const float* cWr[3];
    for (int i = 0; i < 3; ++i) {
        int base = 18 + 7 * i;
        amW[i] = (const float*)d_in[base + 0];
        amb[i] = (const float*)d_in[base + 1];
        axW[i] = (const float*)d_in[base + 2];
        axb[i] = (const float*)d_in[base + 3];
        cWl[i] = (const float*)d_in[base + 4];
        cbl[i] = (const float*)d_in[base + 5];
        cWr[i] = (const float*)d_in[base + 6];
    }
    const float* fc1W = (const float*)d_in[46];
    const float* fc1b = (const float*)d_in[47];
    const float* fc2W = (const float*)d_in[48];
    const float* fc2b = (const float*)d_in[49];
    float* out = (float*)d_out;

    // ---- workspace layout (floats) ----
    float* ws = (float*)d_ws;
    size_t o = 0;
    float* agg   = ws + o; o += (size_t)16384 * FL;
    float* cnt   = ws + o; o += 16384;
    float* x1    = ws + o; o += (size_t)16384 * FL;
    float* x1p   = ws + o; o += (size_t)4096 * FL;
    float* x2    = ws + o; o += (size_t)4096 * FL;
    float* x2p   = ws + o; o += (size_t)1024 * FL;
    float* x3    = ws + o; o += (size_t)1024 * FL;
    float* x3p   = ws + o; o += (size_t)256 * FL;
    float* x4    = ws + o; o += (size_t)256 * FL;
    float* meanb = ws + o; o += (size_t)4096 * FL;
    float* meanv = ws + o; o += 64;
    float* maxv  = ws + o; o += 64;
    float* attv  = ws + o; o += 64;
    float* chv   = ws + o; o += 4096;
    float* h1    = ws + o; o += 256;
    (void)ws_size;

    auto run_att = [&](float* x, int N, const int* e, int E, int bi) {
        fill_zero<<<gb(N * FL), 256, 0, stream>>>(agg, N * FL);
        fill_zero<<<gb(N), 256, 0, stream>>>(cnt, N);
        scatter_add<<<gb(E * FL), 256, 0, stream>>>(x, e, E, FL, agg, cnt);
        col_reduce<<<FL, 128, 0, stream>>>(x, N, meanv, maxv);
        att_gate<<<1, 64, 0, stream>>>(meanv, maxv, amW[bi], amb[bi], axW[bi], axb[bi], attv);
        ch_gate<<<gb(N), 128, 0, stream>>>(x, agg, cnt, cWl[bi], cbl[bi], cWr[bi], chv, N);
        att_apply<<<gb(N * FL), 256, 0, stream>>>(x, attv, chv, N);
    };

    auto run_sage = [&](const float* xin, float* xout, int N, const int* e, int E, int li) {
        fill_zero<<<gb(N * FL), 256, 0, stream>>>(agg, N * FL);
        fill_zero<<<gb(N), 256, 0, stream>>>(cnt, N);
        scatter_add<<<gb(E * FL), 256, 0, stream>>>(xin, e, E, FL, agg, cnt);
        mean_kernel<<<gb(N * FL), 256, 0, stream>>>(agg, cnt, meanb, N);
        sage_gemm_wmma<<<dim3(N / 16, FL / 16), 32, 0, stream>>>(
            meanb, xin, gWl[li], gWr[li], gbl[li], xout, N, 1);
    };

    // ---- level 1 (C_in = 1, scalar SAGE) ----
    fill_zero<<<gb(16384), 256, 0, stream>>>(agg, 16384);
    fill_zero<<<gb(16384), 256, 0, stream>>>(cnt, 16384);
    scatter_add<<<gb(65536), 256, 0, stream>>>(x_image, e0, 65536, 1, agg, cnt);
    sage1<<<gb(16384 * FL), 256, 0, stream>>>(x_image, agg, cnt, gWl[0], gbl[0], gWr[0], x1);
    pool2<<<gb(4096 * FL), 256, 0, stream>>>(x1, x1p, 64);
    run_att(x1p, 4096, e1, 16384, 0);

    // ---- level 2 ----
    run_sage(x1p, x2, 4096, e1, 16384, 1);
    pool2<<<gb(1024 * FL), 256, 0, stream>>>(x2, x2p, 32);
    run_att(x2p, 1024, e2, 4096, 1);

    // ---- level 3 ----
    run_sage(x2p, x3, 1024, e2, 4096, 2);
    pool2<<<gb(256 * FL), 256, 0, stream>>>(x3, x3p, 16);
    run_att(x3p, 256, e3, 1024, 2);

    // ---- level 4 (no pool / no att) ----
    run_sage(x3p, x4, 256, e3, 1024, 3);

    // ---- head ----
    fc1_kernel<<<256, 128, 0, stream>>>(x4, fc1W, fc1b, h1);
    fc2_logsoftmax<<<1, 32, 0, stream>>>(h1, fc2W, fc2b, out);
}
